// ConditionalMSAWithRoPE_38431367364990
// MI455X (gfx1250) — compile-verified, matrix operand order checked by reading, not run on hardware
//
#include <hip/hip_runtime.h>
#include <hip/hip_bf16.h>
#include <math.h>

// ---------------------------------------------------------------------------
// ConditionalMSAWithRoPE for MI455X (gfx1250, wave32, WMMA f16 16x16x32)
//
// Pipeline:
//   K0: convert w_qkv (3072x1024) f32 -> f16
//   K1: seq_expand (K=8) + RMSNorm, store xA[b][e][d] = norm(x)[b,d,e] as f16
//   K2: qkv = xA * w_qkv^T via WMMA; RoPE fused; epilogue scatters q/k/v
//       DIRECTLY into per-lane WMMA fragment layouts:
//         aq[bh][qtile][kb][lane][16]          (q as A-fragments)
//         kv[bh][blk][frag 0..3][lane][16]     (k as score B-fragments)
//         kv[bh][blk][frag 4..7][lane][16]     (v as PV B-fragments)
//       so attention reads each tile as one contiguous 32B-per-lane load.
//   K3: flash attention per (b,h,128q): K/V fragments async-staged to LDS
//       (global_load_async_to_lds_b128 + s_wait_asynccnt) and shared by all
//       8 waves; scores + online softmax + P*V all via v_wmma_f32_16x16x32_f16.
//   K4: t[b][d]  = sum_n w_seqc[n] * o[b][n][d]
//   K5: out[b][e] = sum_d t[b][d]*w_head[e][d] + b_head[e]*sum(w_seqc)
// ---------------------------------------------------------------------------

typedef __attribute__((ext_vector_type(16))) _Float16 v16h;
typedef __attribute__((ext_vector_type(8)))  _Float16 v8h;
typedef __attribute__((ext_vector_type(8)))  float    v8f;

// Types for the async-copy builtin (param type revealed by clang diagnostic:
// "__attribute__((__vector_size__(4*sizeof(int)))) int __device__ *").
typedef int v4i __attribute__((vector_size(16)));
typedef __attribute__((address_space(1))) v4i as1_v4i;
typedef __attribute__((address_space(3))) v4i as3_v4i;

#define DIM   1024
#define HEADS 16
#define HD    64
#define BATCH 4

#if __has_builtin(__builtin_amdgcn_global_load_async_to_lds_b128)
#define USE_ASYNC_LDS 1
#else
#define USE_ASYNC_LDS 0
#endif

__device__ __forceinline__ v8f wmma_f16(v16h a, v16h b, v8f c) {
  // D = A(16x32 f16) x B(32x16 f16) + C(16x16 f32)
  return __builtin_amdgcn_wmma_f32_16x16x32_f16(
      /*neg_a=*/false, a, /*neg_b=*/false, b,
      /*c_mod=*/(short)0, c, /*reuse_a=*/false, /*reuse_b=*/false);
}

// Fragment element index for the 16-bit 16x32 A / 32x16 B WMMA layout:
//   lane = g*16 + (row|col),  element j:
//     j in [0,8)  -> K = g*8 + j
//     j in [8,16) -> K = g*8 + 16 + (j-8)
// Inverse (K -> (g', j)): g' = (K>>3)&1 ; j = (K&7) + (K>=16 ? 8 : 0)
__device__ __forceinline__ int frag_j(int K) { return (K & 7) + ((K >= 16) ? 8 : 0); }
__device__ __forceinline__ int frag_g(int K) { return (K >> 3) & 1; }

// A-tile (16x32 f16) gathered from row-major memory (used by the qkv GEMM).
__device__ __forceinline__ v16h load_tile_rowmajor(const _Float16* base, int ld, int lane) {
  const int m = lane & 15, g = lane >> 4;
  const _Float16* p = base + (size_t)m * ld + g * 8;
  v8h lo = *(const v8h*)(p);
  v8h hi = *(const v8h*)(p + 16);
  v16h r;
#pragma unroll
  for (int j = 0; j < 8; ++j) { r[j] = lo[j]; r[8 + j] = hi[j]; }
  return r;
}

// Copy 32B global -> LDS, async (ASYNCcnt) when available.
__device__ __forceinline__ void stage32B(const _Float16* g, _Float16* l) {
#if USE_ASYNC_LDS
  __builtin_amdgcn_global_load_async_to_lds_b128(
      (as1_v4i*)g, (as3_v4i*)l, 0, 0);
  __builtin_amdgcn_global_load_async_to_lds_b128(
      (as1_v4i*)(g + 8), (as3_v4i*)(l + 8), 0, 0);
#else
  ((v8h*)l)[0] = ((const v8h*)g)[0];
  ((v8h*)l)[1] = ((const v8h*)g)[1];
#endif
}

__device__ __forceinline__ void wait_async_stage() {
#if USE_ASYNC_LDS
#if __has_builtin(__builtin_amdgcn_s_wait_asynccnt)
  __builtin_amdgcn_s_wait_asynccnt(0);
#else
  asm volatile("s_wait_asynccnt 0" ::: "memory");
#endif
#endif
}

// ---------------------------------------------------------------------------
// K0: f32 -> f16 convert
__global__ __launch_bounds__(256) void cvt_f16_kernel(const float* __restrict__ src,
                                                      _Float16* __restrict__ dst, int n) {
  int i = blockIdx.x * 256 + threadIdx.x;
  if (i < n) dst[i] = (_Float16)src[i];
}

// ---------------------------------------------------------------------------
// K1: one block per (b,d): x[b,d,e] = sum_s emb[b,s,d]*w_se[e,s] + b_se[e],
// RMSNorm over e, store transposed xA[b][e][d] (f16).
__global__ __launch_bounds__(256) void expand_norm_kernel(
    const float* __restrict__ emb, const float* __restrict__ wse,
    const float* __restrict__ bse, const float* __restrict__ nw,
    _Float16* __restrict__ xA) {
  const int bd = blockIdx.x;
  const int b = bd >> 10, d = bd & 1023;
  __shared__ float red[256];

  float e8[8];
#pragma unroll
  for (int s = 0; s < 8; ++s) e8[s] = emb[(size_t)(b * 8 + s) * DIM + d];

  float vals[4], ss = 0.f;
#pragma unroll
  for (int j = 0; j < 4; ++j) {
    const int e = threadIdx.x + j * 256;
    const float* wr = wse + (size_t)e * 8;
    float a = bse[e];
#pragma unroll
    for (int s = 0; s < 8; ++s) a = fmaf(e8[s], wr[s], a);
    vals[j] = a;
    ss = fmaf(a, a, ss);
  }
  red[threadIdx.x] = ss;
  __syncthreads();
  for (int off = 128; off > 0; off >>= 1) {
    if (threadIdx.x < off) red[threadIdx.x] += red[threadIdx.x + off];
    __syncthreads();
  }
  const float r = rsqrtf(red[0] * (1.f / DIM) + 1e-6f);
#pragma unroll
  for (int j = 0; j < 4; ++j) {
    const int e = threadIdx.x + j * 256;
    xA[((size_t)b * DIM + e) * DIM + d] = (_Float16)(vals[j] * r * nw[e]);
  }
}

// ---------------------------------------------------------------------------
// K2: qkv GEMM via WMMA + RoPE, scattering into WMMA fragment layouts.
// Grid: b(4) x ntile(64) x egroup(6); block = 256 threads = 8 waves.
__global__ __launch_bounds__(256) void qkv_rope_kernel(
    const _Float16* __restrict__ xA, const _Float16* __restrict__ wq,
    _Float16* __restrict__ aq, _Float16* __restrict__ kv) {
  const int lane = threadIdx.x & 31;
  const int w    = threadIdx.x >> 5;
  const int eg   = blockIdx.x % 6;
  const int nt   = (blockIdx.x / 6) & 63;
  const int b    = blockIdx.x / (6 * 64);
  const int e0   = eg * 512 + w * 64;

  const _Float16* xb = xA + (size_t)b * DIM * DIM + (size_t)nt * 16 * DIM;

  v8f acc0 = {}, acc1 = {}, acc2 = {}, acc3 = {};
  for (int k0 = 0; k0 < DIM; k0 += 32) {
    v16h a = load_tile_rowmajor(xb + k0, DIM, lane);
    __builtin_prefetch(xb + k0 + 32, 0, 1);
    // B[K][n] = wq[e0+n][k0+K] : K-contiguous per lane row
    v16h b0 = load_tile_rowmajor(wq + (size_t)(e0 +  0) * DIM + k0, DIM, lane);
    v16h b1 = load_tile_rowmajor(wq + (size_t)(e0 + 16) * DIM + k0, DIM, lane);
    v16h b2 = load_tile_rowmajor(wq + (size_t)(e0 + 32) * DIM + k0, DIM, lane);
    v16h b3 = load_tile_rowmajor(wq + (size_t)(e0 + 48) * DIM + k0, DIM, lane);
    acc0 = wmma_f16(a, b0, acc0);
    acc1 = wmma_f16(a, b1, acc1);
    acc2 = wmma_f16(a, b2, acc2);
    acc3 = wmma_f16(a, b3, acc3);
  }

  const int g = lane >> 4, col = lane & 15;
  v8f accs[4] = {acc0, acc1, acc2, acc3};
#pragma unroll
  for (int c = 0; c < 4; ++c) {
    const int ecol = e0 + c * 16 + col;
#pragma unroll
    for (int i = 0; i < 8; ++i) {
      const int pos = nt * 16 + g * 8 + i;
      float v = accs[c][i];
      if (e0 < 2 * DIM) {
        // q or k: RoPE. Pair partner (dd^1) lives in lane^1 of the C-tile.
        const float part = __shfl_xor(v, 1, 32);
        const int dd = ecol & 63;
        const float fr = (float)pos *
            __expf(-(float)(dd & ~1) * (9.210340371976184f / 64.f)); // 10000^(-(dd&~1)/64)
        const float cs = __cosf(fr), sn = __sinf(fr);
        v = (dd & 1) ? fmaf(part, sn, v * cs) : fmaf(-part, sn, v * cs);
        const int h  = (ecol >> 6) & 15;
        const int bh = b * HEADS + h;
        const int kb = dd >> 5, kl = dd & 31;          // d-halves 0..31 / 32..63
        if (ecol < DIM) {
          // q -> A-fragment: row m = pos%16, K = dd within half
          const int qtile = pos >> 4, m = pos & 15;
          const int lanep = frag_g(kl) * 16 + m;
          aq[((((size_t)bh * 64 + qtile) * 2 + kb) * 32 + lanep) * 16 + frag_j(kl)] =
              (_Float16)v;
        } else {
          // k -> score B-fragment: col n = pos%16, K = dd within half
          const int blk = pos >> 5, kt2 = (pos >> 4) & 1, n = pos & 15;
          const int lanep = frag_g(kl) * 16 + n;
          const int frag  = kt2 * 2 + kb;              // frags 0..3
          kv[((((size_t)bh * 32 + blk) * 8 + frag) * 32 + lanep) * 16 + frag_j(kl)] =
              (_Float16)v;
        }
      } else {
        // v -> PV B-fragment: K = pos within 32-block, col n = dd%16
        const int ev = ecol - 2 * DIM;
        const int h = ev >> 6, dd = ev & 63;
        const int bh = b * HEADS + h;
        const int blk = pos >> 5, kl = pos & 31;
        const int c2 = dd >> 4, n = dd & 15;
        const int lanep = frag_g(kl) * 16 + n;
        const int frag  = 4 + c2;                      // frags 4..7
        kv[((((size_t)bh * 32 + blk) * 8 + frag) * 32 + lanep) * 16 + frag_j(kl)] =
            (_Float16)v;
      }
    }
  }
}

// ---------------------------------------------------------------------------
// K3: flash attention. Grid = 64(bh) * 8(qgroup); block = 256 = 8 waves.
// Per 32-key block: 8KB of K/V fragments async-staged into LDS, consumed by
// all 8 waves. All tile reads are contiguous 32B per lane.
__global__ __launch_bounds__(256) void attention_kernel(
    const _Float16* __restrict__ aq, const _Float16* __restrict__ kv,
    float* __restrict__ o) {
  const int lane = threadIdx.x & 31;
  const int w    = threadIdx.x >> 5;
  const int bh   = blockIdx.x >> 3;
  const int qgrp = blockIdx.x & 7;
  const int b = bh >> 4, h = bh & 15;
  const int qtile = qgrp * 8 + w;
  const int qbase = qtile * 16;

  __shared__ __align__(32) _Float16 kvstage[8 * 512]; // 8 fragments = 8KB
  __shared__ __align__(32) _Float16 pstage[8 * 512];  // per-wave P A-frag, 8KB
  _Float16* pw = &pstage[w * 512];

  // Q A-fragments (loaded once): contiguous 32B per lane.
  const _Float16* ap = aq + (((size_t)bh * 64 + qtile) * 2) * 512;
  v16h a0 = *(const v16h*)(ap + (size_t)lane * 16);
  v16h a1 = *(const v16h*)(ap + 512 + (size_t)lane * 16);

  float mrow[8], lrow[8];
  v8f ov0 = {}, ov1 = {}, ov2 = {}, ov3 = {};
#pragma unroll
  for (int i = 0; i < 8; ++i) { mrow[i] = -3.0e38f; lrow[i] = 0.f; }

  const int g = lane >> 4, col = lane & 15;
  const int plane = ((col >> 3) & 1) * 16;   // lane-group of P fragment target
  const int pj = col & 7;

  for (int blk = 0; blk < 32; ++blk) {
    // ---- cooperative async stage of 8 fragments (4 K + 4 V) into LDS
    const _Float16* src =
        kv + ((size_t)bh * 32 + blk) * 8 * 512 + (size_t)threadIdx.x * 16;
    stage32B(src, &kvstage[threadIdx.x * 16]);
    wait_async_stage();
    __syncthreads();

    // ---- scores: kpos tiles [blk*32, +16) and [+16, +32)
    v16h k00 = *(const v16h*)(&kvstage[0 * 512 + lane * 16]); // kt2=0, d 0..31
    v16h k01 = *(const v16h*)(&kvstage[1 * 512 + lane * 16]); // kt2=0, d 32..63
    v16h k10 = *(const v16h*)(&kvstage[2 * 512 + lane * 16]); // kt2=1, d 0..31
    v16h k11 = *(const v16h*)(&kvstage[3 * 512 + lane * 16]); // kt2=1, d 32..63
    v8f s0 = {}, s1 = {};
    s0 = wmma_f16(a0, k00, s0);
    s0 = wmma_f16(a1, k01, s0);
    s1 = wmma_f16(a0, k10, s1);
    s1 = wmma_f16(a1, k11, s1);

    // ---- online softmax (row reductions within 16-lane half-groups)
    float alpha[8];
#pragma unroll
    for (int i = 0; i < 8; ++i) {
      float v0 = s0[i] * 0.125f;   // 1/sqrt(64)
      float v1 = s1[i] * 0.125f;
      float rm = fmaxf(v0, v1);
#pragma unroll
      for (int msk = 1; msk < 16; msk <<= 1) rm = fmaxf(rm, __shfl_xor(rm, msk, 16));
      const float mnew = fmaxf(mrow[i], rm);
      alpha[i] = __expf(mrow[i] - mnew);
      const float p0 = __expf(v0 - mnew);
      const float p1 = __expf(v1 - mnew);
      float rs = p0 + p1;
#pragma unroll
      for (int msk = 1; msk < 16; msk <<= 1) rs += __shfl_xor(rs, msk, 16);
      lrow[i] = lrow[i] * alpha[i] + rs;
      mrow[i] = mnew;
      s0[i] = p0;
      s1[i] = p1;
    }
#pragma unroll
    for (int i = 0; i < 8; ++i) {
      ov0[i] *= alpha[i]; ov1[i] *= alpha[i];
      ov2[i] *= alpha[i]; ov3[i] *= alpha[i];
    }

    // ---- write P directly in A-fragment order to LDS
#pragma unroll
    for (int i = 0; i < 8; ++i) {
      const int qm = g * 8 + i;                 // P row
      pw[(plane + qm) * 16 + pj]     = (_Float16)s0[i];  // K = col      (<16)
      pw[(plane + qm) * 16 + pj + 8] = (_Float16)s1[i];  // K = 16 + col
    }
    asm volatile("s_wait_dscnt 0" ::: "memory");
    v16h pa = *(const v16h*)(pw + (size_t)lane * 16);

    // ---- O += P(16x32) x V(32x16) for 4 dd-chunks
    v16h bv0 = *(const v16h*)(&kvstage[4 * 512 + lane * 16]);
    v16h bv1 = *(const v16h*)(&kvstage[5 * 512 + lane * 16]);
    v16h bv2 = *(const v16h*)(&kvstage[6 * 512 + lane * 16]);
    v16h bv3 = *(const v16h*)(&kvstage[7 * 512 + lane * 16]);
    ov0 = wmma_f16(pa, bv0, ov0);
    ov1 = wmma_f16(pa, bv1, ov1);
    ov2 = wmma_f16(pa, bv2, ov2);
    ov3 = wmma_f16(pa, bv3, ov3);

    __syncthreads();   // all waves done with kvstage before restage
  }

  // ---- finalize: divide by l, scatter to o[b][pos][h*64+dd] (f32)
#pragma unroll
  for (int i = 0; i < 8; ++i) {
    const float inv = 1.f / lrow[i];
    const int pos = qbase + g * 8 + i;
    float* orow = o + ((size_t)b * DIM + pos) * DIM + h * HD;
    orow[ 0 + col] = ov0[i] * inv;
    orow[16 + col] = ov1[i] * inv;
    orow[32 + col] = ov2[i] * inv;
    orow[48 + col] = ov3[i] * inv;
  }
}

// ---------------------------------------------------------------------------
// K4: t[b][d] = sum_n w_seqc[n] * o[b][n][d]
__global__ __launch_bounds__(256) void reduce_o_kernel(
    const float* __restrict__ o, const float* __restrict__ wsc, float* __restrict__ t) {
  const int b = blockIdx.x >> 2;
  const int d = (blockIdx.x & 3) * 256 + threadIdx.x;
  const float* ob = o + (size_t)b * DIM * DIM + d;
  float acc = 0.f;
  for (int n = 0; n < DIM; ++n) acc = fmaf(wsc[n], ob[(size_t)n * DIM], acc);
  t[b * DIM + d] = acc;
}

// ---------------------------------------------------------------------------
// K5: out[b][e] = sum_d t[b][d]*w_head[e][d] + b_head[e]*sum(w_seqc)
__global__ __launch_bounds__(256) void head_out_kernel(
    const float* __restrict__ t, const float* __restrict__ wh,
    const float* __restrict__ bh, const float* __restrict__ wsc,
    float* __restrict__ out) {
  const int b = blockIdx.x >> 2;
  const int e = (blockIdx.x & 3) * 256 + threadIdx.x;
  float cs = 0.f;
  for (int n = 0; n < DIM; ++n) cs += wsc[n];
  const float* tb = t + b * DIM;
  const float* wr = wh + (size_t)e * DIM;
  float acc = 0.f;
  for (int d = 0; d < DIM; ++d) acc = fmaf(tb[d], wr[d], acc);
  out[b * DIM + e] = acc + bh[e] * cs;
}

// ---------------------------------------------------------------------------
extern "C" void kernel_launch(void* const* d_in, const int* in_sizes, int n_in,
                              void* d_out, int out_size, void* d_ws, size_t ws_size,
                              hipStream_t stream) {
  (void)in_sizes; (void)n_in; (void)out_size; (void)ws_size;
  const float* emb  = (const float*)d_in[0];  // (4,8,1024)
  const float* wse  = (const float*)d_in[1];  // (1024,8)
  const float* bse  = (const float*)d_in[2];  // (1024,)
  const float* nw   = (const float*)d_in[3];  // (1024,)
  const float* wqkv = (const float*)d_in[4];  // (3072,1024)
  const float* wh   = (const float*)d_in[5];  // (1024,1024)
  const float* bh   = (const float*)d_in[6];  // (1024,)
  const float* wsc  = (const float*)d_in[7];  // (1,1024)
  float* out = (float*)d_out;                 // (4,1024)

  // workspace layout (bytes)
  char* ws = (char*)d_ws;
  _Float16* xA   = (_Float16*)(ws);                  // 8 MiB
  _Float16* wqh  = (_Float16*)(ws + (8u << 20));     // 6 MiB
  _Float16* aq   = (_Float16*)(ws + (14u << 20));    // 8 MiB  (q A-fragments)
  _Float16* kv   = (_Float16*)(ws + (22u << 20));    // 16 MiB (k+v B-fragments)
  float*    oBuf = (float*)   (ws + (38u << 20));    // 16 MiB
  float*    tBuf = (float*)   (ws + (54u << 20));    // 16 KiB

  cvt_f16_kernel<<<(3072 * 1024) / 256, 256, 0, stream>>>(wqkv, wqh, 3072 * 1024);
  expand_norm_kernel<<<BATCH * DIM, 256, 0, stream>>>(emb, wse, bse, nw, xA);
  qkv_rope_kernel<<<BATCH * 64 * 6, 256, 0, stream>>>(xA, wqh, aq, kv);
  attention_kernel<<<BATCH * HEADS * 8, 256, 0, stream>>>(aq, kv, oBuf);
  reduce_o_kernel<<<BATCH * 4, 256, 0, stream>>>(oBuf, wsc, tBuf);
  head_out_kernel<<<BATCH * 4, 256, 0, stream>>>(tBuf, wh, bh, wsc, out);
}
